// SelfAttention_88029649698817
// MI455X (gfx1250) — matrix operand order repaired and checked by reading, not compiled, not run on hardware
//
#include <hip/hip_runtime.h>

typedef __attribute__((ext_vector_type(16))) _Float16 v16h;
typedef __attribute__((ext_vector_type(8)))  float    v8f;
typedef __attribute__((ext_vector_type(4)))  float    v4f;
typedef __attribute__((ext_vector_type(2)))  float    f32x2;
typedef __attribute__((ext_vector_type(2)))  _Float16 h16x2;
typedef __attribute__((ext_vector_type(4)))  int      i32x4;

#define B_  4
#define S_  2048
#define H_  16
#define D_  64
#define IN_ 1024

#define AS1 __attribute__((address_space(1)))
#define AS3 __attribute__((address_space(3)))

#if defined(__has_builtin)
#if __has_builtin(__builtin_amdgcn_global_load_async_to_lds_b128)
#define HAVE_ASYNC 1
#endif
#endif
#ifndef HAVE_ASYNC
#define HAVE_ASYNC 0
#warning "gfx1250 __builtin_amdgcn_global_load_async_to_lds_b128 not available; falling back to synchronous LDS staging"
#endif

union FragU {
  v16h v;
  v4f  f4[2];
  _Float16 e[16];
};

// ---- fragment helpers (wave32, v_wmma_f32_16x16x32_f16) ----
// A-matrix 16x32 f16 (ISA 7.12.2): lane row m = lane&15;
//   here lanes 0-15 hold K 0..15, lanes 16-31 hold K 16..31 (contiguous order chosen by this code, not the ISA table order)
__device__ inline v16h load_afrag(const _Float16* rowspan32, int sel) {
  const v4f* p = (const v4f*)rowspan32;
  FragU u;
  u.f4[0] = p[2 * sel];
  u.f4[1] = p[2 * sel + 1];
  return u.v;
}
// B-matrix 32x16 f16: lane col n = lane&15; lanes 0-15 hold K=0..15, lanes 16-31 K=16..31
__device__ inline v16h load_bfrag(const _Float16* kspan32, int sel) {
  const v4f* p = (const v4f*)kspan32;
  FragU u;
  u.f4[0] = p[2 * sel];
  u.f4[1] = p[2 * sel + 1];
  return u.v;
}
__device__ inline v8f wmma_f16(v16h a, v16h b, v8f c) {
  return __builtin_amdgcn_wmma_f32_16x16x32_f16(false, a, false, b, (short)0, c,
                                                false, false);
}
__device__ inline v8f zero8() {
  v8f z = {0.f,0.f,0.f,0.f,0.f,0.f,0.f,0.f};
  return z;
}

// 16-byte global -> LDS copy (async DMA path when available)
__device__ inline void async_copy16(const _Float16* g, _Float16* l) {
#if HAVE_ASYNC
  __builtin_amdgcn_global_load_async_to_lds_b128(
      (AS1 i32x4*)(_Float16*)g, (AS3 i32x4*)l, 0, 0);
#else
  *(v4f*)l = *(const v4f*)g;
#endif
}

template<int N> __device__ inline void wait_async() {
#if HAVE_ASYNC
#if __has_builtin(__builtin_amdgcn_s_wait_asynccnt)
  __builtin_amdgcn_s_wait_asynccnt(N);
#else
  asm volatile("s_wait_asynccnt %0" :: "i"(N) : "memory");
#endif
#endif
}

// ---- kernel 0: pack weights into B-fragment-native f16 layout ----
// WqkvP: [mat 3][head 16][chunk 2][nb 4][lane 32][16]
// WoP  : [chunk 32][nb 4][lane 32][16]
__global__ void __launch_bounds__(256)
pack_weights(const float* __restrict__ Wq, const float* __restrict__ Wk,
             const float* __restrict__ Wv, const float* __restrict__ Wo,
             _Float16* __restrict__ WqkvP, _Float16* __restrict__ WoP) {
  int id = blockIdx.x * blockDim.x + threadIdx.x;
  const int NQKV = 3 * 16 * 2 * 4 * 32 * 16;   // 196608
  const int NO   = 32 * 4 * 32 * 16;           // 65536
  if (id < NQKV) {
    int e  = id & 15;
    int l  = (id >> 4) & 31;
    int nb = (id >> 9) & 3;
    int c  = (id >> 11) & 1;
    int h  = (id >> 12) & 15;
    int m  = id >> 16;
    int k   = c * 32 + ((l & 16) ? 16 : 0) + e;     // B-layout: K = klo + e
    int col = h * 64 + nb * 16 + (l & 15);
    const float* W = (m == 0) ? Wq : (m == 1) ? Wk : Wv;
    WqkvP[id] = (_Float16)W[k * IN_ + col];
  } else if (id < NQKV + NO) {
    int id2 = id - NQKV;
    int e  = id2 & 15;
    int l  = (id2 >> 4) & 31;
    int nb = (id2 >> 9) & 3;
    int c  = id2 >> 11;
    int k = c * 32 + ((l & 16) ? 16 : 0) + e;
    int n = nb * 16 + (l & 15);
    WoP[id2] = (_Float16)Wo[k * D_ + n];
  }
}

// ---- kernel 1: Q,K,V projections (one wave per (b,h,128 rows)) ----
// Qh,Kh: [b][h][s][64] f16 row-major ; Vt: [b][h][64][s] f16 (d-major)
// C-tiles bounce through wave-private LDS so global stores are coalesced b128s.
__global__ void __launch_bounds__(256)
qkv_proj(const float* __restrict__ x, const _Float16* __restrict__ WqkvP,
         _Float16* __restrict__ Qh, _Float16* __restrict__ Kh,
         _Float16* __restrict__ Vt) {
  __shared__ __align__(16) _Float16 Stg[8][1024];   // 2KB per wave
  int tid  = threadIdx.x;
  int lane = tid & 31;
  int sel  = lane >> 4;
  int ln   = lane & 15;
  int wv   = tid >> 5;
  int w    = blockIdx.x * 8 + wv;
  int st128 = w & 15;
  int h     = (w >> 4) & 15;
  int b     = w >> 8;
  int rowBase = st128 * 128;
  const float* xb = x + (size_t)b * S_ * D_;
  _Float16* st = Stg[wv];

  for (int stile = 0; stile < 8; ++stile) {
    int r0 = rowBase + stile * 16;
    const float* xrow = xb + (size_t)(r0 + ln) * D_;
    v16h a[2];
#pragma unroll
    for (int c = 0; c < 2; ++c) {
      FragU u;
#pragma unroll
      for (int i = 0; i < 8; ++i) {
        int k = c * 32 + 2 * i + sel * 16;
        f32x2 f = *(const f32x2*)(xrow + k);
        u.e[2 * i]     = (_Float16)f.x;
        u.e[2 * i + 1] = (_Float16)f.y;
      }
      a[c] = u.v;
    }
#pragma unroll
    for (int m = 0; m < 3; ++m) {
      v8f acc[4];
#pragma unroll
      for (int nb = 0; nb < 4; ++nb) {
        acc[nb] = zero8();
#pragma unroll
        for (int c = 0; c < 2; ++c) {
          const _Float16* bp =
              WqkvP + ((((size_t)(m * 16 + h) * 2 + c) * 4 + nb) * 32 + lane) * 16;
          FragU u;
          const v4f* p4 = (const v4f*)bp;
          u.f4[0] = p4[0]; u.f4[1] = p4[1];
          acc[nb] = wmma_f16(a[c], u.v, acc[nb]);
        }
      }
      if (m < 2) {
        // stage 16x64 row-major tile, flush 2KB contiguous
#pragma unroll
        for (int nb = 0; nb < 4; ++nb)
#pragma unroll
          for (int i = 0; i < 8; ++i)
            st[(i + sel * 8) * 64 + nb * 16 + ln] = (_Float16)acc[nb][i];
        _Float16* gdst = (m == 0 ? Qh : Kh) + (((size_t)b * H_ + h) * S_ + r0) * D_;
        const v4f* sp = (const v4f*)(st + lane * 32);
        v4f* dp = (v4f*)(gdst + lane * 32);
#pragma unroll
        for (int j = 0; j < 4; ++j) dp[j] = sp[j];
      } else {
        // stage transposed 64x16 tile, flush 2 rows per lane (b128 each half)
#pragma unroll
        for (int nb = 0; nb < 4; ++nb)
#pragma unroll
          for (int i = 0; i < 8; i += 2) {
            h16x2 pk;
            pk.x = (_Float16)acc[nb][i];
            pk.y = (_Float16)acc[nb][i + 1];
            *(h16x2*)(st + (nb * 16 + ln) * 16 + i + sel * 8) = pk;
          }
        _Float16* vbase = Vt + ((size_t)b * H_ + h) * D_ * (size_t)S_ + r0;
#pragma unroll
        for (int rr = 0; rr < 2; ++rr) {
          int d = lane + rr * 32;
          const v4f* sp = (const v4f*)(st + d * 16);
          v4f* dp = (v4f*)(vbase + (size_t)d * S_);
          dp[0] = sp[0]; dp[1] = sp[1];
        }
      }
    }
  }
}

// ---- kernel 2: flash attention ----
// One block = one (b,h) x 128 query rows (8 waves x 16 rows).
// K/V tiles staged cooperatively into double-buffered LDS with async-to-LDS,
// pipelined on ASYNCcnt; per-wave causal compute skipping (wave-uniform branch).
__global__ void __launch_bounds__(256)
attn(const _Float16* __restrict__ Qh, const _Float16* __restrict__ Kh,
     const _Float16* __restrict__ Vt, _Float16* __restrict__ Oh,
     const int* __restrict__ cmaskp) {
  __shared__ __align__(16) _Float16 ldsK[2][32 * 64];  // [buf][key][d]
  __shared__ __align__(16) _Float16 ldsV[2][64 * 32];  // [buf][d][key]
  __shared__ __align__(16) _Float16 Pbuf[8 * 16 * 32]; // wave-private P bounce

  int tid  = threadIdx.x;
  int lane = tid & 31;
  int sel  = lane >> 4;
  int ln   = lane & 15;
  int wv   = tid >> 5;
  int bid  = blockIdx.x;
  int qb   = bid & 15;          // 128-row query block within S
  int h    = (bid >> 4) & 15;
  int b    = bid >> 8;
  int q0   = qb * 128 + wv * 16;
  int causal = cmaskp[0];

  const _Float16* Qb = Qh + ((size_t)b * H_ + h) * S_ * (size_t)D_;
  const _Float16* Kb = Kh + ((size_t)b * H_ + h) * S_ * (size_t)D_;
  const _Float16* Vb = Vt + ((size_t)b * H_ + h) * D_ * (size_t)S_;
  _Float16* Plds = Pbuf + wv * (16 * 32);

  const _Float16* qrow = Qb + (size_t)(q0 + ln) * D_;
  v16h qa0 = load_afrag(qrow, sel);
  v16h qa1 = load_afrag(qrow + 32, sel);

  v8f acc[4];
#pragma unroll
  for (int nb = 0; nb < 4; ++nb) acc[nb] = zero8();
  float rm[8], rl[8];
#pragma unroll
  for (int i = 0; i < 8; ++i) { rm[i] = -1e30f; rl[i] = 0.0f; }

  // per-thread share of an 8KB (K+V) tile: 2 x b128
  int vr = tid >> 2, vs = tid & 3;
  auto stage = [&](int kt, int buf) {
    int k0 = kt * 32;
    const _Float16* gK = Kb + (size_t)k0 * D_ + tid * 8;
    const _Float16* gV = Vb + (size_t)vr * S_ + k0 + vs * 8;
    async_copy16(gK, &ldsK[buf][tid * 8]);
    async_copy16(gV, &ldsV[buf][vr * 32 + vs * 8]);
  };

  const int KT = causal ? (qb * 4 + 4) : (S_ / 32);  // block-uniform tile count
  stage(0, 0);
  for (int kt = 0; kt < KT; ++kt) {
    int cur = kt & 1;
    bool more = (kt + 1 < KT);
    if (more) stage(kt + 1, cur ^ 1);
    if (more) wait_async<2>(); else wait_async<0>();
    __syncthreads();                 // tile kt visible to all waves

    int k0 = kt * 32;
    if (!causal || k0 <= q0 + 15) {  // wave-uniform: EXEC stays all-ones
      const _Float16* Kl = ldsK[cur];
      const _Float16* Vl = ldsV[cur];
      v8f s[2];
#pragma unroll
      for (int nsub = 0; nsub < 2; ++nsub) {
        const _Float16* krow = Kl + (size_t)(nsub * 16 + ln) * D_;
        v16h kb0 = load_bfrag(krow, sel);
        v16h kb1 = load_bfrag(krow + 32, sel);
        v8f z = zero8();
        z = wmma_f16(qa0, kb0, z);
        z = wmma_f16(qa1, kb1, z);
        s[nsub] = z;
      }
      // scale by 1/sqrt(64), causal mask
#pragma unroll
      for (int nsub = 0; nsub < 2; ++nsub) {
        int kj = k0 + nsub * 16 + ln;
#pragma unroll
        for (int i = 0; i < 8; ++i) {
          int qi = q0 + i + sel * 8;
          float v = s[nsub][i] * 0.125f;
          s[nsub][i] = (causal && (kj > qi)) ? -1e30f : v;
        }
      }
      // online softmax (xor reductions stay within 16-lane halves)
      float nm[8], fsc[8];
#pragma unroll
      for (int i = 0; i < 8; ++i) {
        float t = fmaxf(s[0][i], s[1][i]);
#pragma unroll
        for (int msk = 8; msk >= 1; msk >>= 1)
          t = fmaxf(t, __shfl_xor(t, msk, 32));
        nm[i]  = fmaxf(rm[i], t);
        fsc[i] = __expf(rm[i] - nm[i]);
        rm[i]  = nm[i];
      }
#pragma unroll
      for (int i = 0; i < 8; ++i) {
        float p0 = __expf(s[0][i] - nm[i]);
        float p1 = __expf(s[1][i] - nm[i]);
        s[0][i] = p0; s[1][i] = p1;
        float t = p0 + p1;
#pragma unroll
        for (int msk = 8; msk >= 1; msk >>= 1)
          t += __shfl_xor(t, msk, 32);
        rl[i] = rl[i] * fsc[i] + t;
#pragma unroll
        for (int nb = 0; nb < 4; ++nb) acc[nb][i] *= fsc[i];
      }
      // P: C-layout -> A-layout via wave-private LDS bounce
#pragma unroll
      for (int i = 0; i < 8; ++i) {
        int r = i + sel * 8;
        Plds[r * 32 + ln]      = (_Float16)s[0][i];
        Plds[r * 32 + 16 + ln] = (_Float16)s[1][i];
      }
      v16h pa = load_afrag(Plds + ln * 32, sel);
#pragma unroll
      for (int nb = 0; nb < 4; ++nb) {
        const _Float16* vrow = Vl + (size_t)(nb * 16 + ln) * 32;
        v16h vb = load_bfrag(vrow, sel);
        acc[nb] = wmma_f16(pa, vb, acc[nb]);
      }
    }
    __syncthreads();                 // all waves done with buf[cur]
  }

  // epilogue: normalize, stage 16x64 O tile in (now dead) K/V LDS, flush b128s
  _Float16* ob = (wv < 4) ? (&ldsK[0][0] + wv * 1024)
                          : (&ldsV[0][0] + (wv - 4) * 1024);
#pragma unroll
  for (int i = 0; i < 8; ++i) {
    float inv = 1.0f / rl[i];
#pragma unroll
    for (int nb = 0; nb < 4; ++nb)
      ob[(i + sel * 8) * 64 + nb * 16 + ln] = (_Float16)(acc[nb][i] * inv);
  }
  {
    int r  = lane >> 1;        // row 0..15
    int hh = lane & 1;         // which 32-half chunk of the row
    const v4f* sp = (const v4f*)(ob + r * 64 + hh * 32);
    v4f* dp = (v4f*)(Oh + ((size_t)b * S_ + q0 + r) * IN_ + h * D_ + hh * 32);
#pragma unroll
    for (int j = 0; j < 4; ++j) dp[j] = sp[j];
  }
}

// ---- kernel 3: out = Oh[B*S,1024] @ Wo + bo ----
__global__ void __launch_bounds__(256)
out_proj(const _Float16* __restrict__ Oh, const _Float16* __restrict__ WoP,
         const float* __restrict__ bo, float* __restrict__ out) {
  __shared__ __align__(16) float Fbuf[8][1024];   // 4KB per wave
  int tid  = threadIdx.x;
  int lane = tid & 31;
  int sel  = lane >> 4;
  int ln   = lane & 15;
  int wv   = tid >> 5;
  int w    = blockIdx.x * 8 + wv;
  int r0   = w * 16;
  const _Float16* arow = Oh + (size_t)(r0 + ln) * IN_;
  v8f acc[4];
#pragma unroll
  for (int nb = 0; nb < 4; ++nb) acc[nb] = zero8();
  for (int c = 0; c < 32; ++c) {
    v16h a = load_afrag(arow + c * 32, sel);
#pragma unroll
    for (int nb = 0; nb < 4; ++nb) {
      const _Float16* bp = WoP + ((size_t)(c * 4 + nb) * 32 + lane) * 16;
      FragU u;
      const v4f* p4 = (const v4f*)bp;
      u.f4[0] = p4[0]; u.f4[1] = p4[1];
      acc[nb] = wmma_f16(a, u.v, acc[nb]);
    }
  }
  // stage 16x64 f32 tile (4KB contiguous in global), flush coalesced b128s
  float* fb = Fbuf[wv];
#pragma unroll
  for (int nb = 0; nb < 4; ++nb) {
    float bias = bo[nb * 16 + ln];
#pragma unroll
    for (int i = 0; i < 8; ++i)
      fb[(i + sel * 8) * 64 + nb * 16 + ln] = acc[nb][i] + bias;
  }
  {
    const v4f* sp = (const v4f*)fb;
    v4f* dp = (v4f*)(out + (size_t)r0 * 64);
#pragma unroll
    for (int j = 0; j < 8; ++j) dp[lane * 8 + j] = sp[lane * 8 + j];
  }
}

extern "C" void kernel_launch(void* const* d_in, const int* in_sizes, int n_in,
                              void* d_out, int out_size, void* d_ws, size_t ws_size,
                              hipStream_t stream) {
  (void)in_sizes; (void)n_in; (void)out_size; (void)ws_size;
  const float* x   = (const float*)d_in[0];
  const float* Wq  = (const float*)d_in[1];
  const float* Wk  = (const float*)d_in[2];
  const float* Wv  = (const float*)d_in[3];
  const float* Wo  = (const float*)d_in[4];
  const float* bo  = (const float*)d_in[5];
  const int* cmask = (const int*)d_in[6];

  char* ws = (char*)d_ws;
  size_t off = 0;
  _Float16* WqkvP = (_Float16*)(ws + off); off += (size_t)3 * 16 * 2 * 4 * 32 * 16 * 2;
  _Float16* WoP   = (_Float16*)(ws + off); off += (size_t)32 * 4 * 32 * 16 * 2;
  _Float16* Qh    = (_Float16*)(ws + off); off += (size_t)B_ * H_ * S_ * D_ * 2;
  _Float16* Kh    = (_Float16*)(ws + off); off += (size_t)B_ * H_ * S_ * D_ * 2;
  _Float16* Vt    = (_Float16*)(ws + off); off += (size_t)B_ * H_ * S_ * D_ * 2;
  _Float16* Oh    = (_Float16*)(ws + off); off += (size_t)B_ * S_ * IN_ * 2;

  pack_weights<<<1024, 256, 0, stream>>>(Wq, Wk, Wv, Wo, WqkvP, WoP);
  qkv_proj<<<128, 256, 0, stream>>>(x, WqkvP, Qh, Kh, Vt);
  attn<<<1024, 256, 0, stream>>>(Qh, Kh, Vt, Oh, cmask);
  out_proj<<<64, 256, 0, stream>>>(Oh, WoP, bo, (float*)d_out);
}
